// CausalSelfAttention_29695403885069
// MI455X (gfx1250) — matrix operand order repaired and checked by reading, not compiled, not run on hardware
//
#include <hip/hip_runtime.h>
#include <hip/hip_bf16.h>

#define B_  2
#define T_  2048
#define C_  1024
#define NH_ 16
#define HS_ 64

typedef __attribute__((ext_vector_type(16))) __bf16 bf16x16;
typedef __attribute__((ext_vector_type(8)))  float  f32x8;

union Frag {
  bf16x16 v;
  unsigned int d[8];
  unsigned short u[16];
};

// fp32 -> bf16 (round to nearest even), bit-level so it also parses on host
__device__ inline unsigned short f2bf(float f) {
  unsigned int u = __builtin_bit_cast(unsigned int, f);
  u += 0x7FFFu + ((u >> 16) & 1u);
  return (unsigned short)(u >> 16);
}

__device__ inline unsigned int pack2(float a, float b) {
  return (unsigned int)f2bf(a) | ((unsigned int)f2bf(b) << 16);
}

__device__ inline f32x8 wmma_bf16(const Frag& a, const Frag& b, f32x8 c) {
  // (neg_a, A, neg_b, B, c_mod, C, reuse_a, reuse_b)
  return __builtin_amdgcn_wmma_f32_16x16x32_bf16(false, a.v, false, b.v, (short)0, c, false, false);
}

// LDS row padding: 48 shorts = 96 B (multiple of 16 B -> b128-aligned rows)
#define LPAD 48

// ---------------------------------------------------------------------------
// Kernel 1: qkv = x @ W_attn + b_attn ; scatter to head-major q (pre-scaled),
// k, and transposed v (vT[b,h][hs][t]) as bf16.
// 128x128 block tile, 8 waves x (16 rows x 128 cols), double-buffered LDS.
// ---------------------------------------------------------------------------
__global__ __launch_bounds__(256) void qkv_gemm_kernel(
    const float* __restrict__ X, const float* __restrict__ W,
    const float* __restrict__ bias,
    unsigned short* __restrict__ q, unsigned short* __restrict__ k,
    unsigned short* __restrict__ vT)
{
  __shared__ __align__(16) unsigned short As[2][128][LPAD];  // [m][k] bf16
  __shared__ __align__(16) unsigned short Bs[2][128][LPAD];  // [n][k] bf16 (B^T)
  const int tid  = threadIdx.x;
  const int wave = tid >> 5, lane = tid & 31, half = lane >> 4, ln = lane & 15;
  const int rowBase = blockIdx.y * 128;
  const int colBase = blockIdx.x * 128;

  // per-thread staging registers for one 128x32 A tile + 32x128 B tile
  float4 xa[4], wb[4];
  const int ar  = (tid >> 3) & 127;          // placeholder (recomputed below)

  f32x8 acc[8];
#pragma unroll
  for (int t = 0; t < 8; ++t) acc[t] = (f32x8)0.0f;

  // ---- tile load helpers (inlined manually) ----
  // A tile: 128 rows x 32 k as 1024 float4; thread -> 4 of them
  // B tile: 32 k x 128 cols as 1024 float4; thread -> 4 of them
#define LOAD_TILES(KK)                                                          \
  {                                                                             \
    _Pragma("unroll")                                                           \
    for (int it = 0; it < 4; ++it) {                                            \
      int v4 = it * 256 + tid;                                                  \
      int r = v4 >> 3, c4 = v4 & 7;                                             \
      xa[it] = *(const float4*)&X[(rowBase + r) * C_ + (KK) + c4 * 4];          \
    }                                                                           \
    _Pragma("unroll")                                                           \
    for (int it = 0; it < 4; ++it) {                                            \
      int v4 = it * 256 + tid;                                                  \
      int kr = v4 >> 5, nc4 = v4 & 31;                                          \
      wb[it] = *(const float4*)&W[((KK) + kr) * (3 * C_) + colBase + nc4 * 4];  \
    }                                                                           \
  }

#define STORE_TILES(BUF)                                                        \
  {                                                                             \
    _Pragma("unroll")                                                           \
    for (int it = 0; it < 4; ++it) {                                            \
      int v4 = it * 256 + tid;                                                  \
      int r = v4 >> 3, c4 = v4 & 7;                                             \
      uint2 p = {pack2(xa[it].x, xa[it].y), pack2(xa[it].z, xa[it].w)};         \
      *(uint2*)&As[BUF][r][c4 * 4] = p;                                         \
    }                                                                           \
    _Pragma("unroll")                                                           \
    for (int it = 0; it < 4; ++it) {                                            \
      int v4 = it * 256 + tid;                                                  \
      int kr = v4 >> 5, nc4 = v4 & 31;                                          \
      Bs[BUF][nc4 * 4 + 0][kr] = f2bf(wb[it].x);                                \
      Bs[BUF][nc4 * 4 + 1][kr] = f2bf(wb[it].y);                                \
      Bs[BUF][nc4 * 4 + 2][kr] = f2bf(wb[it].z);                                \
      Bs[BUF][nc4 * 4 + 3][kr] = f2bf(wb[it].w);                                \
    }                                                                           \
  }

  LOAD_TILES(0)
  STORE_TILES(0)
  __syncthreads();

  int cur = 0;
  for (int kk0 = 0; kk0 < C_; kk0 += 32, cur ^= 1) {
    const bool haveNext = (kk0 + 32) < C_;
    if (haveNext) LOAD_TILES(kk0 + 32)       // global loads in flight over WMMAs

    Frag a;   // A operand: lane m = ln; element e -> k = half*16 + e (linear)
#pragma unroll
    for (int r = 0; r < 8; ++r)
      a.d[r] = *(const unsigned int*)&As[cur][wave * 16 + ln]
                                         [half * 16 + r * 2];
#pragma unroll
    for (int t = 0; t < 8; ++t) {
      Frag b; // B operand: lane n = ln; element e -> k = half*16 + e (linear)
#pragma unroll
      for (int r = 0; r < 8; ++r)
        b.d[r] = *(const unsigned int*)&Bs[cur][t * 16 + ln][half * 16 + r * 2];
      acc[t] = wmma_bf16(a, b, acc[t]);
    }

    if (haveNext) STORE_TILES(cur ^ 1)
    __syncthreads();
  }

  // Epilogue. Whole block lies in exactly one of q/k/v: uniform branch.
  const int seg = blockIdx.x >> 3;           // 0=q, 1=k, 2=v
  const int colLoc = colBase - (seg << 10);
#pragma unroll
  for (int t = 0; t < 8; ++t) {
#pragma unroll
    for (int e = 0; e < 8; ++e) {
      int col = colBase + t * 16 + ln;       // in [0,3C)
      int c2  = colLoc + t * 16 + ln;        // in [0,C)
      int row = rowBase + wave * 16 + e + half * 8;
      float val = acc[t][e] + bias[col];
      int bb = row >> 11, tq = row & (T_ - 1);
      int h = c2 >> 6, hs = c2 & 63;
      if (seg == 0) {
        q[((bb * NH_ + h) * T_ + tq) * HS_ + hs] = f2bf(val * 0.125f);
      } else if (seg == 1) {
        k[((bb * NH_ + h) * T_ + tq) * HS_ + hs] = f2bf(val);
      } else {
        vT[((bb * NH_ + h) * HS_ + hs) * T_ + tq] = f2bf(val);
      }
    }
  }
#undef LOAD_TILES
#undef STORE_TILES
}

// ---------------------------------------------------------------------------
// Kernel 2: flash attention. One wave per (b,h,16-row q tile), 4 waves/block.
// ---------------------------------------------------------------------------
__global__ __launch_bounds__(128) void attn_kernel(
    const unsigned short* __restrict__ qb, const unsigned short* __restrict__ kb,
    const unsigned short* __restrict__ vT, unsigned short* __restrict__ y)
{
  __shared__ __align__(16) unsigned short P[4][16][LPAD]; // per-wave P tile
  const int wave = threadIdx.x >> 5, lane = threadIdx.x & 31;
  const int half = lane >> 4, ln = lane & 15;
  const int wid = blockIdx.x * 4 + wave;
  const int bh  = wid >> 7;                 // 128 q-tiles per (b,h)
  const int q0  = (wid & 127) * 16;
  const int bIdx = bh >> 4, h = bh & 15;
  const unsigned short* Q = qb + (size_t)bh * T_ * HS_;
  const unsigned short* K = kb + (size_t)bh * T_ * HS_;
  const unsigned short* V = vT + (size_t)bh * HS_ * T_;   // [hs][t]

  Frag qa[2];  // Q rows as A-operand, K-dim = HS split into 2x32 (reg-resident)
#pragma unroll
  for (int kk = 0; kk < 2; ++kk)
#pragma unroll
    for (int r = 0; r < 8; ++r)
      qa[kk].d[r] = *(const unsigned int*)&Q[(q0 + ln) * HS_ + kk * 32 +
                                             half * 16 + r * 2];

  f32x8 o[4];
#pragma unroll
  for (int t = 0; t < 4; ++t) o[t] = (f32x8)0.0f;
  float mrow[8], lrow[8];
#pragma unroll
  for (int e = 0; e < 8; ++e) { mrow[e] = -1e30f; lrow[e] = 0.0f; }

  for (int k0 = 0; k0 < q0 + 16; k0 += 32) {        // 32 keys per step
    f32x8 s0 = (f32x8)0.0f, s1 = (f32x8)0.0f;
#pragma unroll
    for (int kk = 0; kk < 2; ++kk) {
      Frag b0, b1;  // B operand = K rows (keys as N), 32 contiguous B per lane
#pragma unroll
      for (int r = 0; r < 8; ++r) {
        b0.d[r] = *(const unsigned int*)&K[(k0 + ln) * HS_ + kk * 32 + half * 16 + r * 2];
        b1.d[r] = *(const unsigned int*)&K[(k0 + 16 + ln) * HS_ + kk * 32 + half * 16 + r * 2];
      }
      s0 = wmma_bf16(qa[kk], b0, s0);
      s1 = wmma_bf16(qa[kk], b1, s1);
    }

    // Hoist V fragment loads: independent of softmax -> overlap with VALU below
    Frag vb[4];
#pragma unroll
    for (int t = 0; t < 4; ++t)
#pragma unroll
      for (int r = 0; r < 8; ++r)
        vb[t].d[r] = *(const unsigned int*)&V[(t * 16 + ln) * T_ + k0 + half * 16 + r * 2];

    if (k0 + 32 > q0) {   // diagonal block: causal mask
#pragma unroll
      for (int e = 0; e < 8; ++e) {
        int row = q0 + e + half * 8;
        if (k0 + ln > row)      s0[e] = -1e30f;
        if (k0 + 16 + ln > row) s1[e] = -1e30f;
      }
    }

#pragma unroll
    for (int e = 0; e < 8; ++e) {   // row m = e + 8*half, spread over 16 lanes
      float mx = fmaxf(s0[e], s1[e]);
      mx = fmaxf(mx, __shfl_xor(mx, 1));
      mx = fmaxf(mx, __shfl_xor(mx, 2));
      mx = fmaxf(mx, __shfl_xor(mx, 4));
      mx = fmaxf(mx, __shfl_xor(mx, 8));
      float mnew = fmaxf(mrow[e], mx);
      float al = __expf(mrow[e] - mnew);
      mrow[e] = mnew;
      float p0 = __expf(s0[e] - mnew);
      float p1 = __expf(s1[e] - mnew);
      float rs = p0 + p1;
      rs += __shfl_xor(rs, 1);
      rs += __shfl_xor(rs, 2);
      rs += __shfl_xor(rs, 4);
      rs += __shfl_xor(rs, 8);
      lrow[e] = lrow[e] * al + rs;
#pragma unroll
      for (int t = 0; t < 4; ++t) o[t][e] *= al;
      P[wave][e + half * 8][ln]      = f2bf(p0);
      P[wave][e + half * 8][16 + ln] = f2bf(p1);
    }

    // same-wave LDS is in-order; fence compiler + DS pipe before re-reading
    asm volatile("s_wait_dscnt 0" ::: "memory");

    Frag pa;  // P as A-operand (16 rows x 32 keys)
#pragma unroll
    for (int r = 0; r < 8; ++r)
      pa.d[r] = *(const unsigned int*)&P[wave][ln]
                                        [half * 16 + r * 2];
#pragma unroll
    for (int t = 0; t < 4; ++t)
      o[t] = wmma_bf16(pa, vb[t], o[t]);
  }

#pragma unroll
  for (int e = 0; e < 8; ++e) {
    float inv = 1.0f / lrow[e];
    int row = q0 + e + half * 8;
#pragma unroll
    for (int t = 0; t < 4; ++t)
      y[(size_t)(bIdx * T_ + row) * C_ + h * HS_ + t * 16 + ln] = f2bf(o[t][e] * inv);
  }
}

// ---------------------------------------------------------------------------
// Kernel 3: out = y @ W_proj + b_proj (f32 output), double-buffered LDS
// ---------------------------------------------------------------------------
__global__ __launch_bounds__(256) void proj_gemm_kernel(
    const unsigned short* __restrict__ Yb, const float* __restrict__ W,
    const float* __restrict__ bias, float* __restrict__ out)
{
  __shared__ __align__(16) unsigned short As[2][128][LPAD];
  __shared__ __align__(16) unsigned short Bs[2][128][LPAD];
  const int tid  = threadIdx.x;
  const int wave = tid >> 5, lane = tid & 31, half = lane >> 4, ln = lane & 15;
  const int rowBase = blockIdx.y * 128;
  const int colBase = blockIdx.x * 128;

  uint4  ya[2];     // 2 x 8 bf16 per thread = A tile (already bf16)
  float4 wb[4];     // B tile staged as f32

  f32x8 acc[8];
#pragma unroll
  for (int t = 0; t < 8; ++t) acc[t] = (f32x8)0.0f;

#define LOAD_TILES2(KK)                                                         \
  {                                                                             \
    _Pragma("unroll")                                                           \
    for (int it = 0; it < 2; ++it) {                                            \
      int v8 = it * 256 + tid;                                                  \
      int r = v8 >> 2, c8 = v8 & 3;                                             \
      ya[it] = *(const uint4*)&Yb[(rowBase + r) * C_ + (KK) + c8 * 8];          \
    }                                                                           \
    _Pragma("unroll")                                                           \
    for (int it = 0; it < 4; ++it) {                                            \
      int v4 = it * 256 + tid;                                                  \
      int kr = v4 >> 5, nc4 = v4 & 31;                                          \
      wb[it] = *(const float4*)&W[((KK) + kr) * C_ + colBase + nc4 * 4];        \
    }                                                                           \
  }

#define STORE_TILES2(BUF)                                                       \
  {                                                                             \
    _Pragma("unroll")                                                           \
    for (int it = 0; it < 2; ++it) {                                            \
      int v8 = it * 256 + tid;                                                  \
      int r = v8 >> 2, c8 = v8 & 3;                                             \
      *(uint4*)&As[BUF][r][c8 * 8] = ya[it];                                    \
    }                                                                           \
    _Pragma("unroll")                                                           \
    for (int it = 0; it < 4; ++it) {                                            \
      int v4 = it * 256 + tid;                                                  \
      int kr = v4 >> 5, nc4 = v4 & 31;                                          \
      Bs[BUF][nc4 * 4 + 0][kr] = f2bf(wb[it].x);                                \
      Bs[BUF][nc4 * 4 + 1][kr] = f2bf(wb[it].y);                                \
      Bs[BUF][nc4 * 4 + 2][kr] = f2bf(wb[it].z);                                \
      Bs[BUF][nc4 * 4 + 3][kr] = f2bf(wb[it].w);                                \
    }                                                                           \
  }

  LOAD_TILES2(0)
  STORE_TILES2(0)
  __syncthreads();

  int cur = 0;
  for (int kk0 = 0; kk0 < C_; kk0 += 32, cur ^= 1) {
    const bool haveNext = (kk0 + 32) < C_;
    if (haveNext) LOAD_TILES2(kk0 + 32)

    Frag a;
#pragma unroll
    for (int r = 0; r < 8; ++r)
      a.d[r] = *(const unsigned int*)&As[cur][wave * 16 + ln]
                                         [half * 16 + r * 2];
#pragma unroll
    for (int t = 0; t < 8; ++t) {
      Frag b;
#pragma unroll
      for (int r = 0; r < 8; ++r)
        b.d[r] = *(const unsigned int*)&Bs[cur][t * 16 + ln][half * 16 + r * 2];
      acc[t] = wmma_bf16(a, b, acc[t]);
    }

    if (haveNext) STORE_TILES2(cur ^ 1)
    __syncthreads();
  }

#pragma unroll
  for (int t = 0; t < 8; ++t) {
#pragma unroll
    for (int e = 0; e < 8; ++e) {
      int col = colBase + t * 16 + ln;
      int row = rowBase + wave * 16 + e + half * 8;
      out[(size_t)row * C_ + col] = acc[t][e] + bias[col];
    }
  }
#undef LOAD_TILES2
#undef STORE_TILES2
}

// ---------------------------------------------------------------------------
extern "C" void kernel_launch(void* const* d_in, const int* in_sizes, int n_in,
                              void* d_out, int out_size, void* d_ws, size_t ws_size,
                              hipStream_t stream) {
  const float* x  = (const float*)d_in[0];   // [B,T,C]
  const float* Wa = (const float*)d_in[1];   // [C,3C]
  const float* ba = (const float*)d_in[2];   // [3C]
  const float* Wp = (const float*)d_in[3];   // [C,C]
  const float* bp = (const float*)d_in[4];   // [C]
  float* out = (float*)d_out;                // [B,T,C] f32

  // Workspace layout (bf16): q(8MB) | k(8MB) | vT(8MB) | y(8MB) = 32MB total
  const size_t QKV_BYTES = (size_t)B_ * NH_ * T_ * HS_ * sizeof(unsigned short);
  char* ws = (char*)d_ws;
  unsigned short* qb = (unsigned short*)(ws);
  unsigned short* kb = (unsigned short*)(ws + QKV_BYTES);
  unsigned short* vT = (unsigned short*)(ws + 2 * QKV_BYTES);
  unsigned short* yb = (unsigned short*)(ws + 3 * QKV_BYTES);

  qkv_gemm_kernel<<<dim3(3 * C_ / 128, (B_ * T_) / 128), 256, 0, stream>>>(
      x, Wa, ba, qb, kb, vT);
  attn_kernel<<<dim3((B_ * NH_ * (T_ / 16)) / 4), 128, 0, stream>>>(
      qb, kb, vT, yb);
  proj_gemm_kernel<<<dim3(C_ / 128, (B_ * T_) / 128), 256, 0, stream>>>(
      yb, Wp, bp, out);
}
